// NetworkObj_48670569398458
// MI455X (gfx1250) — matrix operand order repaired and checked by reading, not compiled, not run on hardware
//
#include <hip/hip_runtime.h>
#include <hip/hip_bf16.h>

typedef _Float16 h16;
typedef __attribute__((ext_vector_type(16))) _Float16 v16h;
typedef __attribute__((ext_vector_type(8)))  _Float16 v8h;
typedef __attribute__((ext_vector_type(8)))  float    v8f;

// ---------------------------------------------------------------------------
// prep: split (B*N,6) f32 input into contiguous coords f32 (B*N,3) and feats
// f16 (B*N,3)
// ---------------------------------------------------------------------------
__global__ void prep_kernel(const float* __restrict__ in, float* __restrict__ coords,
                            h16* __restrict__ feats, int total) {
  int i = blockIdx.x * 256 + threadIdx.x;
  if (i >= total) return;
  const float* p = in + (size_t)i * 6;
  coords[(size_t)i * 3 + 0] = p[0];
  coords[(size_t)i * 3 + 1] = p[1];
  coords[(size_t)i * 3 + 2] = p[2];
  feats[(size_t)i * 3 + 0] = (h16)p[3];
  feats[(size_t)i * 3 + 1] = (h16)p[4];
  feats[(size_t)i * 3 + 2] = (h16)p[5];
}

// ---------------------------------------------------------------------------
// pack W (Kr x Nr f32 row-major) -> B fragments for v_wmma_f32_16x16x32_f16.
// Layout: [ntile][ktile][lane(32)][elem(16)] halves, zero padded to (Kp,Np).
// Lane L holds column N = ntile*16 + (L&15), K = ktile*32 + (L>>4)*8 + (elem&7) + (elem>>3)*16.
// ---------------------------------------------------------------------------
__global__ void pack_w_kernel(const float* __restrict__ W, h16* __restrict__ Bp,
                              int Kp, int Np, int Kr, int Nr) {
  size_t i = (size_t)blockIdx.x * 256 + threadIdx.x;
  size_t total = (size_t)Kp * Np;
  if (i >= total) return;
  int e    = (int)(i & 15);
  int lane = (int)((i >> 4) & 31);
  size_t rest = i >> 9;
  int nkt = Kp >> 5;
  int kt = (int)(rest % nkt);
  int nt = (int)(rest / nkt);
  int nn = nt * 16 + (lane & 15);
  int kk = kt * 32 + (lane >> 4) * 8 + (e & 7) + (e >> 3) * 16;
  float v = (kk < Kr && nn < Nr) ? W[(size_t)kk * Nr + nn] : 0.0f;
  Bp[i] = (h16)v;
}

// ---------------------------------------------------------------------------
// Farthest point sampling: one block (256 threads) per batch, LDS min-dist.
// Matches jnp.argmax first-index tie-break. Also gathers new_xyz.
// ---------------------------------------------------------------------------
__global__ void fps_kernel(const float* __restrict__ coords, int N, int S,
                           int* __restrict__ fidx, float* __restrict__ nxyz) {
  __shared__ float md[2048];
  __shared__ float rv[256];
  __shared__ int   ri[256];
  int b = blockIdx.x, t = threadIdx.x;
  const float* base = coords + (size_t)b * N * 3;
  for (int n = t; n < N; n += 256) md[n] = 1e10f;
  __syncthreads();
  int far = 0;
  for (int it = 0; it < S; ++it) {
    if (t == 0) fidx[b * S + it] = far;
    float cx = base[far * 3 + 0], cy = base[far * 3 + 1], cz = base[far * 3 + 2];
    float bv = -1.0f; int bi = 0x7fffffff;
    for (int n = t; n < N; n += 256) {
      float dx = base[n * 3 + 0] - cx;
      float dy = base[n * 3 + 1] - cy;
      float dz = base[n * 3 + 2] - cz;
      float d = dx * dx + dy * dy + dz * dz;
      float m = md[n]; m = fminf(m, d); md[n] = m;
      if (m > bv) { bv = m; bi = n; }
    }
    rv[t] = bv; ri[t] = bi;
    __syncthreads();
    for (int o = 128; o > 0; o >>= 1) {
      if (t < o) {
        float ov = rv[t + o]; int oi = ri[t + o];
        if (ov > rv[t] || (ov == rv[t] && oi < ri[t])) { rv[t] = ov; ri[t] = oi; }
      }
      __syncthreads();
    }
    far = ri[0];
    __syncthreads();
  }
  for (int s = t; s < S; s += 256) {
    int id = fidx[b * S + s];
    nxyz[(size_t)(b * S + s) * 3 + 0] = base[id * 3 + 0];
    nxyz[(size_t)(b * S + s) * 3 + 1] = base[id * 3 + 1];
    nxyz[(size_t)(b * S + s) * 3 + 2] = base[id * 3 + 2];
  }
}

// ---------------------------------------------------------------------------
// Ball query + grouping. One block (64 threads) per centroid. First 64 indices
// with d^2 <= r^2 in ascending index order (== reference's sort trick), padded
// with the first hit. Writes f16 rows [rel_xyz(3) | feats(Cf) | zeros -> Kpad].
// ---------------------------------------------------------------------------
__global__ void group_kernel(const float* __restrict__ coords, const h16* __restrict__ feats,
                             const float* __restrict__ nxyz, h16* __restrict__ A,
                             int N, int S, int Cf, int Kpad, float r2) {
  __shared__ int lidx[64];
  int bs = blockIdx.x;
  int b = bs / S;
  const float* cb = coords + (size_t)b * N * 3;
  float cx = nxyz[(size_t)bs * 3 + 0];
  float cy = nxyz[(size_t)bs * 3 + 1];
  float cz = nxyz[(size_t)bs * 3 + 2];
  if (threadIdx.x == 0) {
    int cnt = 0;
    for (int n = 0; n < N && cnt < 64; ++n) {
      float dx = cb[n * 3 + 0] - cx;
      float dy = cb[n * 3 + 1] - cy;
      float dz = cb[n * 3 + 2] - cz;
      if (dx * dx + dy * dy + dz * dz <= r2) lidx[cnt++] = n;
    }
    int f = (cnt > 0) ? lidx[0] : 0;
    for (int i = cnt; i < 64; ++i) lidx[i] = f;
  }
  __syncthreads();
  int t = threadIdx.x;
  int gi = lidx[t];
  h16* arow = A + ((size_t)bs * 64 + t) * Kpad;
  arow[0] = (h16)(cb[gi * 3 + 0] - cx);
  arow[1] = (h16)(cb[gi * 3 + 1] - cy);
  arow[2] = (h16)(cb[gi * 3 + 2] - cz);
  const h16* fr = feats + ((size_t)b * N + gi) * Cf;
  for (int c = 0; c < Cf; ++c) arow[3 + c] = fr[c];
  for (int c = 3 + Cf; c < Kpad; ++c) arow[c] = (h16)0.0f;
}

// group_all for SA3: rows = b*128+p, [xyz(3) | pts(C) | zeros -> Kpad]
__global__ void groupall_kernel(const float* __restrict__ nxyz2, const h16* __restrict__ P2,
                                h16* __restrict__ A3, int C, int Kpad) {
  int row = blockIdx.x;
  int t = threadIdx.x;
  h16* arow = A3 + (size_t)row * Kpad;
  if (t < 3) arow[t] = (h16)nxyz2[(size_t)row * 3 + t];
  for (int c = t; c < C; c += 64) arow[3 + c] = P2[(size_t)row * C + c];
  for (int c = 3 + C + t; c < Kpad; c += 64) arow[c] = (h16)0.0f;
}

// ---------------------------------------------------------------------------
// WMMA GEMM: Y[R,N] = A[R,K] @ Wpacked + bias. 4 waves per block, each wave a
// 16x16 tile via v_wmma_f32_16x16x32_f16. A fragment loaded per ISA 16-bit A
// layout (row = lane&15; lane<16 holds K {0..7,16..23}, lane>=16 {8..15,24..31})
// as two contiguous 16B loads. B fragment is one contiguous 32B load.
// ---------------------------------------------------------------------------
__global__ void gemm_wmma_kernel(const h16* __restrict__ A, const h16* __restrict__ Bp,
                                 const float* __restrict__ bias, int biasN,
                                 h16* __restrict__ Yh, float* __restrict__ Yf,
                                 int R, int K, int N) {
  const int lane = threadIdx.x & 31;
  const int wave = threadIdx.x >> 5;
  const int m0 = (blockIdx.x * 4 + wave) * 16;
  if (m0 >= R) return;
  const int ntile = blockIdx.y;
  const int halfsel = lane >> 4;
  const int row = m0 + (lane & 15);
  const int kb = halfsel * 8;
  const h16* arow = A + (size_t)row * K;
  const h16* bptr = Bp + ((size_t)ntile * (K >> 5) * 32 + lane) * 16;
  v8f acc = {};
  for (int kt = 0; kt < K; kt += 32) {
    union { v16h v; v8h h[2]; } au;
    au.h[0] = *(const v8h*)(arow + kt + kb);
    au.h[1] = *(const v8h*)(arow + kt + kb + 16);
    v16h bfrag = *(const v16h*)bptr;
    __builtin_prefetch((const void*)(bptr + 512), 0, 1);  // next K-tile of B
    bptr += 512;
    acc = __builtin_amdgcn_wmma_f32_16x16x32_f16(false, au.v, false, bfrag,
                                                 (short)0, acc, false, false);
  }
  const int col = ntile * 16 + (lane & 15);
  float bv = (col < biasN) ? bias[col] : 0.0f;
  const int rbase = m0 + halfsel * 8;  // C layout: vgpr i -> M = i (+8 for hi lanes)
#pragma unroll
  for (int i = 0; i < 8; ++i) {
    float v = acc[i] + bv;
    size_t o = (size_t)(rbase + i) * N + col;
    if (Yh) Yh[o] = (h16)v;
    if (Yf) Yf[o] = v;
  }
}

// ---------------------------------------------------------------------------
// Deterministic per-column sum / sumsq (two stage, fixed 2048-row blocks)
// ---------------------------------------------------------------------------
__global__ void colsum_partial_kernel(const h16* __restrict__ Y, float* __restrict__ partial,
                                      int R, int N) {
  int c = blockIdx.x * 64 + threadIdx.x;
  int rb = blockIdx.y;
  long r0 = (long)rb * 2048;
  long r1 = r0 + 2048; if (r1 > R) r1 = R;
  float s = 0.0f, q = 0.0f;
  const h16* p = Y + (size_t)r0 * N + c;
  for (long r = r0; r < r1; ++r) { float v = (float)*p; s += v; q += v * v; p += N; }
  partial[(size_t)rb * N * 2 + c] = s;
  partial[(size_t)rb * N * 2 + N + c] = q;
}

__global__ void colsum_final_kernel(const float* __restrict__ partial, float* __restrict__ stats,
                                    int N, int nrb) {
  int c = blockIdx.x * 64 + threadIdx.x;
  float s = 0.0f, q = 0.0f;
  for (int rb = 0; rb < nrb; ++rb) {
    s += partial[(size_t)rb * N * 2 + c];
    q += partial[(size_t)rb * N * 2 + N + c];
  }
  stats[c] = s;
  stats[N + c] = q;
}

__global__ void bn_relu_kernel(h16* __restrict__ Y, const float* __restrict__ stats,
                               const float* __restrict__ gamma, const float* __restrict__ beta,
                               long total, int N, float invR) {
  long i = (long)blockIdx.x * 256 + threadIdx.x;
  if (i >= total) return;
  int c = (int)(i % N);
  float mean = stats[c] * invR;
  float var  = stats[N + c] * invR - mean * mean;
  float sc = gamma[c] * rsqrtf(var + 1e-5f);
  float sh = beta[c] - mean * sc;
  float v = (float)Y[i] * sc + sh;
  Y[i] = (h16)fmaxf(v, 0.0f);
}

__global__ void maxpool_kernel(const h16* __restrict__ Y, h16* __restrict__ P,
                               int G, int S, int C) {
  long i = (long)blockIdx.x * 256 + threadIdx.x;
  if (i >= (long)G * C) return;
  long g = i / C; int c = (int)(i % C);
  const h16* p = Y + (size_t)g * S * C + c;
  float m = -1e30f;
  for (int s = 0; s < S; ++s) { float v = (float)*p; if (v > m) m = v; p += C; }
  P[i] = (h16)m;
}

__global__ void final_copy_kernel(const float* __restrict__ Yf, float* __restrict__ out,
                                  int B, int Np, int Nr) {
  int i = blockIdx.x * 256 + threadIdx.x;
  if (i >= B * Nr) return;
  int b = i / Nr, c = i % Nr;
  out[i] = Yf[(size_t)b * Np + c];
}

// ---------------------------------------------------------------------------
extern "C" void kernel_launch(void* const* d_in, const int* in_sizes, int n_in,
                              void* d_out, int out_size, void* d_ws, size_t ws_size,
                              hipStream_t stream) {
  (void)in_sizes; (void)n_in; (void)out_size; (void)ws_size;
  const float* xyz = (const float*)d_in[0];

  struct L { int w, b, g, be; int Kp, Np, Kr, Nr; };
  const L Ls[12] = {
    { 1,  2,  3,  4,   32,   64,    6,   64},   // sa1 L0
    { 5,  6,  7,  8,   64,   64,   64,   64},   // sa1 L1
    { 9, 10, 11, 12,   64,  128,   64,  128},   // sa1 L2
    {13, 14, 15, 16,  160,  128,  131,  128},   // sa2 L0
    {17, 18, 19, 20,  128,  128,  128,  128},   // sa2 L1
    {21, 22, 23, 24,  128,  256,  128,  256},   // sa2 L2
    {25, 26, 27, 28,  288,  256,  259,  256},   // sa3 L0
    {29, 30, 31, 32,  256,  512,  256,  512},   // sa3 L1
    {33, 34, 35, 36,  512, 1024,  512, 1024},   // sa3 L2
    {37, 38, 39, 40, 1024,  512, 1024,  512},   // fc1
    {41, 42, 43, 44,  512,  256,  512,  256},   // fc2
    {45, 46, -1, -1,  256,   64,  256,   42},   // fc3 (no BN)
  };

  char* base = (char*)d_ws;
  size_t off = 0;
  auto alloc = [&](size_t bytes) -> void* {
    void* p = base + off;
    off = (off + bytes + 255) & ~(size_t)255;
    return p;
  };

  float* coords = (float*)alloc((size_t)16 * 2048 * 3 * 4);
  h16*   feats0 = (h16*)alloc((size_t)16 * 2048 * 3 * 2);
  int*   fidx1  = (int*)alloc((size_t)16 * 512 * 4);
  float* nxyz1  = (float*)alloc((size_t)16 * 512 * 3 * 4);
  int*   fidx2  = (int*)alloc((size_t)16 * 128 * 4);
  float* nxyz2  = (float*)alloc((size_t)16 * 128 * 3 * 4);
  h16* Wp[12];
  for (int i = 0; i < 12; ++i) Wp[i] = (h16*)alloc((size_t)Ls[i].Kp * Ls[i].Np * 2);
  float* partial = (float*)alloc((size_t)512 * 1024);          // >= max nrb*N*2*4
  float* stats   = (float*)alloc((size_t)2 * 1024 * 4);
  h16* P1  = (h16*)alloc((size_t)8192 * 128 * 2);
  h16* P2  = (h16*)alloc((size_t)2048 * 256 * 2);
  h16* P3  = (h16*)alloc((size_t)16 * 1024 * 2);
  h16* A3  = (h16*)alloc((size_t)2048 * 288 * 2);
  h16* Y3a = (h16*)alloc((size_t)2048 * 256 * 2);
  h16* Y3b = (h16*)alloc((size_t)2048 * 512 * 2);
  h16* Y3c = (h16*)alloc((size_t)2048 * 1024 * 2);
  h16* Yf1 = (h16*)alloc((size_t)16 * 512 * 2);
  h16* Yf2 = (h16*)alloc((size_t)16 * 256 * 2);
  float* Yf3 = (float*)alloc((size_t)16 * 64 * 4);
  // big reusable arenas
  h16* BIG0 = (h16*)alloc((size_t)524288 * 128 * 2);  // SA1 Y3, then SA2 Y2b
  h16* BIG1 = (h16*)alloc((size_t)524288 * 64 * 2);   // SA1 Y1, then SA2 Y2c
  h16* BIG2 = (h16*)alloc((size_t)524288 * 64 * 2);   // SA1 Y2, then SA2 A2
  h16* BIG3 = (h16*)alloc((size_t)524288 * 32 * 2);   // SA1 A1, then SA2 Y2a

  auto gemm = [&](const h16* Ain, int li, h16* Yh, float* Yfo, int R) {
    dim3 g((unsigned)((R + 63) / 64), (unsigned)(Ls[li].Np / 16));
    gemm_wmma_kernel<<<g, 128, 0, stream>>>(Ain, Wp[li], (const float*)d_in[Ls[li].b],
                                            Ls[li].Nr, Yh, Yfo, R, Ls[li].Kp, Ls[li].Np);
  };
  auto bn = [&](h16* Y, int R, int N, int li) {
    int nrb = (R + 2047) / 2048;
    colsum_partial_kernel<<<dim3((unsigned)(N / 64), (unsigned)nrb), 64, 0, stream>>>(Y, partial, R, N);
    colsum_final_kernel<<<(unsigned)(N / 64), 64, 0, stream>>>(partial, stats, N, nrb);
    long tot = (long)R * N;
    bn_relu_kernel<<<(unsigned)((tot + 255) / 256), 256, 0, stream>>>(
        Y, stats, (const float*)d_in[Ls[li].g], (const float*)d_in[Ls[li].be],
        tot, N, 1.0f / (float)R);
  };

  // prep + pack weights (deterministic, re-done every call)
  prep_kernel<<<(16 * 2048 + 255) / 256, 256, 0, stream>>>(xyz, coords, feats0, 16 * 2048);
  for (int i = 0; i < 12; ++i) {
    size_t tot = (size_t)Ls[i].Kp * Ls[i].Np;
    pack_w_kernel<<<(unsigned)((tot + 255) / 256), 256, 0, stream>>>(
        (const float*)d_in[Ls[i].w], Wp[i], Ls[i].Kp, Ls[i].Np, Ls[i].Kr, Ls[i].Nr);
  }

  // ---------------- SA1: N=2048 -> S=512, nsample=64, r=0.2 ----------------
  fps_kernel<<<16, 256, 0, stream>>>(coords, 2048, 512, fidx1, nxyz1);
  h16* A1 = BIG3;
  group_kernel<<<16 * 512, 64, 0, stream>>>(coords, feats0, nxyz1, A1, 2048, 512, 3, 32, 0.04f);
  h16* Y1 = BIG1; h16* Y2 = BIG2; h16* Y3 = BIG0;
  gemm(A1, 0, Y1, nullptr, 524288); bn(Y1, 524288,  64, 0);
  gemm(Y1, 1, Y2, nullptr, 524288); bn(Y2, 524288,  64, 1);
  gemm(Y2, 2, Y3, nullptr, 524288); bn(Y3, 524288, 128, 2);
  maxpool_kernel<<<(8192 * 128 + 255) / 256, 256, 0, stream>>>(Y3, P1, 8192, 64, 128);

  // ---------------- SA2: N=512 -> S=128, nsample=64, r=0.5 -----------------
  fps_kernel<<<16, 256, 0, stream>>>(nxyz1, 512, 128, fidx2, nxyz2);
  h16* A2 = BIG2;
  group_kernel<<<16 * 128, 64, 0, stream>>>(nxyz1, P1, nxyz2, A2, 512, 128, 128, 160, 0.25f);
  h16* Y2a = BIG3; h16* Y2b = BIG0; h16* Y2c = BIG1;
  gemm(A2,  3, Y2a, nullptr, 131072); bn(Y2a, 131072, 128, 3);
  gemm(Y2a, 4, Y2b, nullptr, 131072); bn(Y2b, 131072, 128, 4);
  gemm(Y2b, 5, Y2c, nullptr, 131072); bn(Y2c, 131072, 256, 5);
  maxpool_kernel<<<(2048 * 256 + 255) / 256, 256, 0, stream>>>(Y2c, P2, 2048, 64, 256);

  // ---------------- SA3: group_all over 128 points -------------------------
  groupall_kernel<<<2048, 64, 0, stream>>>(nxyz2, P2, A3, 256, 288);
  gemm(A3,  6, Y3a, nullptr, 2048); bn(Y3a, 2048,  256, 6);
  gemm(Y3a, 7, Y3b, nullptr, 2048); bn(Y3b, 2048,  512, 7);
  gemm(Y3b, 8, Y3c, nullptr, 2048); bn(Y3c, 2048, 1024, 8);
  maxpool_kernel<<<(16 * 1024 + 255) / 256, 256, 0, stream>>>(Y3c, P3, 16, 128, 1024);

  // ---------------- FC head ------------------------------------------------
  gemm(P3,  9, Yf1, nullptr, 16); bn(Yf1, 16, 512, 9);
  gemm(Yf1, 10, Yf2, nullptr, 16); bn(Yf2, 16, 256, 10);
  gemm(Yf2, 11, nullptr, Yf3, 16);  // no BN, f32 logits
  final_copy_kernel<<<(16 * 42 + 255) / 256, 256, 0, stream>>>(Yf3, (float*)d_out, 16, 64, 42);
}
